// Attention_11192684774105
// MI455X (gfx1250) — matrix operand order repaired and checked by reading, not compiled, not run on hardware
//
#include <hip/hip_runtime.h>

// ---------------------------------------------------------------------------
// MI455X (gfx1250) fused attention block, round 3:
//   - S^T via v_wmma_f32_16x16x32_bf16 (lane = query layout): register-local
//     softmax + v_permlanex16 cross-half exchange, zero LDS in the hot loop.
//   - Q pre-scaled by SCALE*log2(e) at qkv time -> WMMA emits scores already
//     in the exp2 domain (no per-iteration scale muls).
//   - raw v_exp_f32 (__builtin_amdgcn_exp2f): no libm range-fixup cmps.
// ---------------------------------------------------------------------------

typedef __attribute__((ext_vector_type(16))) __bf16 v16bf;
typedef __attribute__((ext_vector_type(2)))  __bf16 bf16x2;
typedef __attribute__((ext_vector_type(8)))  float  v8f;

#define N_SP   4096
#define C_CH   128
#define HEADS  2
#define KD     32
#define HD     64
// SCALE * log2(e), folded into Q at qkv time
#define SL2E   0.25501500398230327f

struct BFrag { uint4 lo, hi; };
struct U8 { unsigned int u[8]; };

static __device__ __forceinline__ v16bf frag_cast(uint4 lo, uint4 hi) {
  BFrag t{lo, hi};
  return __builtin_bit_cast(v16bf, t);
}
static __device__ __forceinline__ unsigned short f2bf(float f) {
  unsigned int u = __float_as_uint(f);
  u += 0x7FFFu + ((u >> 16) & 1u);
  return (unsigned short)(u >> 16);
}
static __device__ __forceinline__ unsigned int pack_bf(float a, float b) {
  bf16x2 t;
  t.x = (__bf16)a;                // RNE f32->bf16; pairs fuse to v_cvt_pk_bf16_f32
  t.y = (__bf16)b;
  return __builtin_bit_cast(unsigned int, t);
}
// raw v_exp_f32: args are <= 0; big-negative flushes to 0 (wanted)
static __device__ __forceinline__ float fast_exp2(float x) {
#if __has_builtin(__builtin_amdgcn_exp2f)
  return __builtin_amdgcn_exp2f(x);
#else
  return exp2f(x);
#endif
}
// lane <-> lane^16 exchange, pure VALU (v_permlanex16_b32, identity selectors)
static __device__ __forceinline__ unsigned int xhalf_u(unsigned int v) {
#if __has_builtin(__builtin_amdgcn_permlanex16)
  return __builtin_amdgcn_permlanex16(v, v, 0x76543210u, 0xFEDCBA98u, false, false);
#else
  return (unsigned int)__shfl_xor((int)v, 16);
#endif
}
static __device__ __forceinline__ float xhalf_f(float v) {
  return __builtin_bit_cast(float, xhalf_u(__builtin_bit_cast(unsigned int, v)));
}

// --------------------------- kernel 1: qkv ---------------------------------
__global__ __launch_bounds__(256) void qkv_kernel(
    const float* __restrict__ x, const float* __restrict__ w,
    unsigned short* __restrict__ qt, unsigned short* __restrict__ kt,
    unsigned short* __restrict__ vb, float* __restrict__ vf) {
  const int n  = blockIdx.x * 256 + threadIdx.x;
  const int oc = blockIdx.y;
  const int b  = blockIdx.z;
  const float* xb = x + (size_t)b * C_CH * N_SP + n;
  const float* wr = w + oc * C_CH;           // uniform per block -> scalar path
  float acc = 0.f;
#pragma unroll 8
  for (int c = 0; c < C_CH; ++c) acc += wr[c] * xb[(size_t)c * N_SP];

  const int h = oc >> 7, s = oc & 127;
  if (s < KD) {
    // Q pre-scaled into the exp2 softmax domain
    qt[((size_t)(b * HEADS + h) * N_SP + n) * KD + s] = f2bf(acc * SL2E);
  } else if (s < 2 * KD) {
    kt[((size_t)(b * HEADS + h) * N_SP + n) * KD + (s - KD)] = f2bf(acc);
  } else {
    const int d = s - 2 * KD;
    vb[((size_t)(b * HEADS + h) * HD + d) * N_SP + n] = f2bf(acc);
    vf[((size_t)b * C_CH + h * HD + d) * N_SP + n]    = acc;
  }
}

// ----------------------- kernel 2: flash attention -------------------------
__global__ __launch_bounds__(256) void attn_kernel(
    const unsigned short* __restrict__ qt, const unsigned short* __restrict__ kt,
    const unsigned short* __restrict__ vb, float* __restrict__ zo) {
  const int tid  = threadIdx.x;
  const int wave = tid >> 5, lane = tid & 31;
  const int l15  = lane & 15, lh = lane >> 4;
  const int bh   = blockIdx.y;
  const int b    = bh >> 1, h = bh & 1;
  const int n0   = (blockIdx.x * 8 + wave) * 16;

  // Q as B-fragment (loop invariant): lane = query col, halves split keydim
  const uint4* qrow = (const uint4*)(qt + ((size_t)bh * N_SP + (n0 + l15)) * KD);
  const v16bf qf = frag_cast(qrow[lh], qrow[2 + lh]);

  v8f Oacc[4] = {v8f{}, v8f{}, v8f{}, v8f{}};
  float m_i = -3.0e38f, l_i = 0.f;

  for (int m0 = 0; m0 < N_SP; m0 += 32) {
    // K as A-fragments: lane = key row, halves split keydim (A pattern)
    const uint4* kr0 = (const uint4*)(kt + ((size_t)bh * N_SP + m0 + l15) * KD);
    const uint4* kr1 = (const uint4*)(kt + ((size_t)bh * N_SP + m0 + 16 + l15) * KD);
    const v16bf kf0 = frag_cast(kr0[lh], kr0[2 + lh]);
    const v16bf kf1 = frag_cast(kr1[lh], kr1[2 + lh]);

    const v8f zf = {};
    // S^T tile (exp2 domain): VGPR r = key (r + 8*lh), lane = query (l15)
    v8f S0 = __builtin_amdgcn_wmma_f32_16x16x32_bf16(false, kf0, false, qf,
                                                     (short)0, zf, false, false);
    v8f S1 = __builtin_amdgcn_wmma_f32_16x16x32_bf16(false, kf1, false, qf,
                                                     (short)0, zf, false, false);

    // ---- row (query) max over 32 keys: 16 local + cross-half swap ----
    float mx;
    {
      float t0 = fmaxf(S0[0], S1[0]), t1 = fmaxf(S0[1], S1[1]);
      float t2 = fmaxf(S0[2], S1[2]), t3 = fmaxf(S0[3], S1[3]);
      float t4 = fmaxf(S0[4], S1[4]), t5 = fmaxf(S0[5], S1[5]);
      float t6 = fmaxf(S0[6], S1[6]), t7 = fmaxf(S0[7], S1[7]);
      t0 = fmaxf(t0, t4); t1 = fmaxf(t1, t5); t2 = fmaxf(t2, t6); t3 = fmaxf(t3, t7);
      t0 = fmaxf(t0, t2); t1 = fmaxf(t1, t3);
      mx = fmaxf(t0, t1);
      mx = fmaxf(mx, xhalf_f(mx));                 // other 16 keys
    }
    const float mnew = fmaxf(m_i, mx);
    const float al   = fast_exp2(m_i - mnew);      // first iter: 0

    // ---- P^T = exp2(S - mnew), row sum ----
#pragma unroll
    for (int r = 0; r < 8; ++r) {
      S0[r] = fast_exp2(S0[r] - mnew);
      S1[r] = fast_exp2(S1[r] - mnew);
    }
    float rs;
    {
      float t0 = S0[0] + S1[0], t1 = S0[1] + S1[1];
      float t2 = S0[2] + S1[2], t3 = S0[3] + S1[3];
      float t4 = S0[4] + S1[4], t5 = S0[5] + S1[5];
      float t6 = S0[6] + S1[6], t7 = S0[7] + S1[7];
      t0 += t4; t1 += t5; t2 += t6; t3 += t7;
      t0 += t2; t1 += t3;
      rs = t0 + t1;
      rs += xhalf_f(rs);
    }
    l_i = l_i * al + rs;
    m_i = mnew;

    // ---- assemble P^T B-fragment (lane = query, VGPRs = packed key pairs) ----
    U8 pfw;
#pragma unroll
    for (int c = 0; c < 4; ++c) {
      const unsigned int lo = pack_bf(S0[2 * c], S0[2 * c + 1]);
      const unsigned int hi = pack_bf(S1[2 * c], S1[2 * c + 1]);
      const unsigned int rlo = xhalf_u(lo);
      const unsigned int rhi = xhalf_u(hi);
      pfw.u[c]     = lh ? rhi : lo;   // keys 16*lh + 2c
      pfw.u[4 + c] = lh ? hi  : rlo;  // keys 16*lh + 8 + 2c
    }
    const v16bf pf = __builtin_bit_cast(v16bf, pfw);

    // O rescale by alpha (lane-uniform scalar)
#pragma unroll
    for (int t = 0; t < 4; ++t)
#pragma unroll
      for (int r = 0; r < 8; ++r) Oacc[t][r] *= al;

    // ---- O^T += V(A-frag) x P^T(B-frag): lane = query, VGPR = head-dim ----
#pragma unroll
    for (int t = 0; t < 4; ++t) {
      const uint4* vrow = (const uint4*)(vb + ((size_t)bh * HD + t * 16 + l15) * N_SP + m0);
      const v16bf vfr = frag_cast(vrow[2 * lh], vrow[2 * lh + 1]);
      Oacc[t] = __builtin_amdgcn_wmma_f32_16x16x32_bf16(false, vfr, false, pf,
                                                        (short)0, Oacc[t], false, false);
    }
  }

  const float inv = 1.0f / l_i;
  const int nq = n0 + l15;                          // lane = query -> coalesced
#pragma unroll
  for (int t = 0; t < 4; ++t) {
#pragma unroll
    for (int r = 0; r < 8; ++r) {
      const int c = h * HD + t * 16 + r + 8 * lh;   // head-dim row
      zo[((size_t)b * C_CH + c) * N_SP + nq] = Oacc[t][r] * inv;
    }
  }
}

// --------------------- kernel 3: depthwise 3x3 "pe" ------------------------
__global__ __launch_bounds__(256) void pe_kernel(
    const float* __restrict__ vf, const float* __restrict__ wpe,
    float* __restrict__ zo) {
  const int n = blockIdx.x * 256 + threadIdx.x;
  const int c = blockIdx.y, b = blockIdx.z;
  const int hh = n >> 6, ww = n & 63;
  const float* vrow = vf + ((size_t)b * C_CH + c) * N_SP;
  const float* wk = wpe + c * 9;
  float acc = 0.f;
#pragma unroll
  for (int dy = -1; dy <= 1; ++dy) {
    const int y = hh + dy;
    if ((unsigned)y >= 64u) continue;
#pragma unroll
    for (int dx = -1; dx <= 1; ++dx) {
      const int xq = ww + dx;
      if ((unsigned)xq >= 64u) continue;
      acc += wk[(dy + 1) * 3 + (dx + 1)] * vrow[y * 64 + xq];
    }
  }
  zo[((size_t)b * C_CH + c) * N_SP + n] += acc;
}

// --------------------------- kernel 4: proj --------------------------------
__global__ __launch_bounds__(256) void proj_kernel(
    const float* __restrict__ zo, const float* __restrict__ w,
    float* __restrict__ out) {
  const int n  = blockIdx.x * 256 + threadIdx.x;
  const int oc = blockIdx.y, b = blockIdx.z;
  const float* zb = zo + (size_t)b * C_CH * N_SP + n;
  const float* wr = w + oc * C_CH;
  float acc = 0.f;
#pragma unroll 8
  for (int c = 0; c < C_CH; ++c) acc += wr[c] * zb[(size_t)c * N_SP];
  out[((size_t)b * C_CH + oc) * N_SP + n] = acc;
}

// ---------------------------------------------------------------------------
extern "C" void kernel_launch(void* const* d_in, const int* in_sizes, int n_in,
                              void* d_out, int out_size, void* d_ws, size_t ws_size,
                              hipStream_t stream) {
  (void)in_sizes; (void)n_in; (void)out_size; (void)ws_size;
  const float* x      = (const float*)d_in[0];
  const float* w_qkv  = (const float*)d_in[1];
  const float* w_pe   = (const float*)d_in[2];
  const float* w_proj = (const float*)d_in[3];
  float* out = (float*)d_out;

  char* ws = (char*)d_ws;                           // 24 MB used
  unsigned short* qt = (unsigned short*)(ws);                   // 2 MB  [bh][n][32] bf16 (pre-scaled)
  unsigned short* kt = (unsigned short*)(ws + (2u  << 20));     // 2 MB  [bh][m][32] bf16
  unsigned short* vb = (unsigned short*)(ws + (4u  << 20));     // 4 MB  [bh][d][N]  bf16
  float*          vf = (float*)        (ws + (8u  << 20));      // 8 MB  [b][c][N]   f32
  float*          zo = (float*)        (ws + (16u << 20));      // 8 MB  attn+pe out f32

  qkv_kernel <<<dim3(N_SP / 256, 256, 4),  256, 0, stream>>>(x, w_qkv, qt, kt, vb, vf);
  attn_kernel<<<dim3(N_SP / 128, 8),       256, 0, stream>>>(qt, kt, vb, zo);
  pe_kernel  <<<dim3(N_SP / 256, C_CH, 4), 256, 0, stream>>>(vf, w_pe, zo);
  proj_kernel<<<dim3(N_SP / 256, C_CH, 4), 256, 0, stream>>>(zo, w_proj, out);
}
